// GraphTransformer_Encoder_56169582297734
// MI455X (gfx1250) — matrix operand order repaired and checked by reading, not compiled, not run on hardware
//
#include <hip/hip_runtime.h>
#include <hip/hip_bf16.h>
#include <math.h>

// ---------------------------------------------------------------------------
// GraphTransformer encoder for MI455X (gfx1250).
// - All dense GEMMs on v_wmma_f32_16x16x32_bf16 (wave32, f32 accumulate),
//   operands pre-converted to bf16 in memory so loads feed WMMA directly.
// - Per k-step, ALL NT B-fragments are loaded into distinct registers before
//   the WMMA burst, letting the scheduler overlap load latency with WMMA
//   issue (progressive s_wait_loadcnt instead of 0x0 before each WMMA).
// - Edge attention: 3 streaming passes; Q/K/V stored bf16 (halves gather
//   traffic), softmax stats + scatter accumulation in f32 atomics.
// ---------------------------------------------------------------------------

typedef __attribute__((ext_vector_type(16))) __bf16 v16bf;
typedef __attribute__((ext_vector_type(8)))  __bf16 v8bf;
typedef __attribute__((ext_vector_type(4)))  __bf16 v4bf;
typedef __attribute__((ext_vector_type(8)))  float  v8f;

constexpr int NNODE  = 50000;
constexpr int NEDGE  = 800000;
constexpr int IN_DIM = 1280;
constexpr int DMODEL = 128;
constexpr int NLAYER = 2;
constexpr int FFDIM  = 512;

enum { ACT_NONE = 0, ACT_GELU = 1, ACT_RELU = 2 };

__device__ __forceinline__ unsigned f2key(float f) {
  unsigned b = __float_as_uint(f);
  return b ^ (unsigned)(((int)b >> 31) | 0x80000000);
}
__device__ __forceinline__ float key2f(unsigned k) {
  unsigned b = k ^ ((k & 0x80000000u) ? 0x80000000u : 0xFFFFFFFFu);
  return __uint_as_float(b);
}

// ---------------------------------------------------------------------------
// f32 -> bf16 bulk conversion (n multiple of 4)
// ---------------------------------------------------------------------------
__global__ void f32_to_bf16_kernel(const float* __restrict__ in, __bf16* __restrict__ out,
                                   size_t n)
{
  size_t i = ((size_t)blockIdx.x * blockDim.x + threadIdx.x) * 4;
  if (i >= n) return;
  float4 v = *(const float4*)(in + i);
  v4bf o;
  o[0] = (__bf16)v.x; o[1] = (__bf16)v.y; o[2] = (__bf16)v.z; o[3] = (__bf16)v.w;
  *(v4bf*)(out + i) = o;
}

// ---------------------------------------------------------------------------
// GEMM: out[M,NC] = act(A[M,K] @ W[NC,K]^T + bias), A/W bf16 in memory.
// One wave computes a 16-row x (NT*16)-col strip; K consumed 32/WMMA.
// Optional f32 and/or bf16 outputs.
// ---------------------------------------------------------------------------
template<int K, int NC, int NT, int ACT>
__global__ __launch_bounds__(256)
void gemm_wmma_kernel(const __bf16* __restrict__ A, const __bf16* __restrict__ W,
                      const float* __restrict__ bias, float* __restrict__ out,
                      __bf16* __restrict__ outb, int M)
{
  constexpr int CG = NC / (16 * NT);           // column groups per row-tile
  const int wid   = (blockIdx.x * blockDim.x + threadIdx.x) >> 5;
  const int lane  = threadIdx.x & 31;
  const int total = (M >> 4) * CG;
  if (wid >= total) return;                    // wave-uniform: EXEC stays full
  const int rt  = wid / CG;
  const int cg  = wid % CG;
  const int hi  = lane >> 4;                   // half-wave select
  const int l15 = lane & 15;
  const int row = rt * 16 + l15;               // A row for this lane

  v8f acc[NT];
  #pragma unroll
  for (int t = 0; t < NT; ++t)
    #pragma unroll
    for (int r = 0; r < 8; ++r) acc[t][r] = 0.0f;

  const __bf16* arow = A + (size_t)row * K;

  for (int k0 = 0; k0 < K; k0 += 32) {
    // A fragment 16x32 bf16: K = hi*16 + e
    v16bf af = *(const v16bf*)(arow + k0 + hi * 16);
    if (k0 + 32 < K) __builtin_prefetch(arow + k0 + 32, 0, 3);   // global_prefetch

    // Issue ALL B-fragment loads first (distinct registers -> one long
    // load clause), then the WMMA burst drains them with overlapping waits.
    v16bf bfr[NT];
    #pragma unroll
    for (int t = 0; t < NT; ++t) {
      // B fragment 32x16 bf16: K = hi*16 + e, N = l15; B[k][n] = W[n][k]
      const int col = (cg * NT + t) * 16 + l15;
      bfr[t] = *(const v16bf*)(W + (size_t)col * K + k0 + hi * 16);
    }
    #pragma unroll
    for (int t = 0; t < NT; ++t) {
      acc[t] = __builtin_amdgcn_wmma_f32_16x16x32_bf16(
          false, af, false, bfr[t], (short)0, acc[t], false, false);
    }
  }

  // Epilogue: C/D layout -> M = r + hi*8, N = l15
  #pragma unroll
  for (int t = 0; t < NT; ++t) {
    const int n = (cg * NT + t) * 16 + l15;
    const float bv = bias ? bias[n] : 0.0f;
    #pragma unroll
    for (int r = 0; r < 8; ++r) {
      const int m = rt * 16 + hi * 8 + r;
      float v = acc[t][r] + bv;
      if (ACT == ACT_GELU) v = 0.5f * v * (1.0f + erff(v * 0.7071067811865475f));
      if (ACT == ACT_RELU) v = fmaxf(v, 0.0f);
      if (out)  out[(size_t)m * NC + n]  = v;
      if (outb) outb[(size_t)m * NC + n] = (__bf16)v;
    }
  }
}

// ---------------------------------------------------------------------------
// Edge bias: bias[e] = (rbf(dist) + edge_emb[etype]) . bias_w + bias_b
// ---------------------------------------------------------------------------
__global__ void edge_bias_kernel(const float* __restrict__ dist, const int* __restrict__ etype,
                                 const float* __restrict__ edge_emb, const float* __restrict__ bw,
                                 const float* __restrict__ bb, float* __restrict__ ebias, int E)
{
  int e = blockIdx.x * blockDim.x + threadIdx.x;
  if (e >= E) return;
  const float width = 20.0f / 15.0f;
  float d = dist[e];
  int et = etype[e];
  float s = bb[0];
  #pragma unroll
  for (int r = 0; r < 16; ++r) {
    float u = (d - width * (float)r) / width;
    s += (expf(-u * u) + edge_emb[et * 16 + r]) * bw[r];
  }
  ebias[e] = s;
}

// ---------------------------------------------------------------------------
// Pass 1: per-edge score per head + segment max (monotone-key atomicMax).
// One wave per edge; lane owns channels [4*lane, 4*lane+4); head = lane/8.
// ---------------------------------------------------------------------------
__global__ __launch_bounds__(256)
void edge_score_kernel(const __bf16* __restrict__ Kf, const __bf16* __restrict__ Qf,
                       const float* __restrict__ ebias, const int* __restrict__ src,
                       const int* __restrict__ dst, const int* __restrict__ etype,
                       float* __restrict__ score, unsigned* __restrict__ mkey, int E)
{
  const int e    = (blockIdx.x * blockDim.x + threadIdx.x) >> 5;
  const int lane = threadIdx.x & 31;
  if (e >= E) return;
  const int s = src[e], d = dst[e], et = etype[e];
  v4bf kv = *(const v4bf*)(Kf + (size_t)s * DMODEL + lane * 4);
  v4bf qv = *(const v4bf*)(Qf + (size_t)d * DMODEL + lane * 4);
  float p = (float)kv[0] * (float)qv[0] + (float)kv[1] * (float)qv[1]
          + (float)kv[2] * (float)qv[2] + (float)kv[3] * (float)qv[3];
  p += __shfl_xor(p, 1); p += __shfl_xor(p, 2); p += __shfl_xor(p, 4);   // 8-lane head sum
  const int head = lane >> 3;
  float sc = p * 0.17677669529663687f + ebias[e];                        // 1/sqrt(32)
  if ((lane & 7) == 0) {
    score[(size_t)e * 4 + head] = sc;
    atomicMax(&mkey[(size_t)(d * 4 + et) * 4 + head], f2key(sc));
  }
}

// ---------------------------------------------------------------------------
// Pass 2: ex = exp(score - segmax); accumulate segment denominator.
// ---------------------------------------------------------------------------
__global__ void edge_softmax_kernel(float* __restrict__ score, const unsigned* __restrict__ mkey,
                                    float* __restrict__ den, const int* __restrict__ dst,
                                    const int* __restrict__ etype, int E)
{
  int idx = blockIdx.x * blockDim.x + threadIdx.x;
  if (idx >= E * 4) return;
  int e = idx >> 2, head = idx & 3;
  int d = dst[e], et = etype[e];
  size_t si = (size_t)(d * 4 + et) * 4 + head;
  float ex = expf(score[idx] - key2f(mkey[si]));
  score[idx] = ex;
  atomicAdd(&den[si], ex);
}

// ---------------------------------------------------------------------------
// Pass 3: out[dst] += (ex/den) * V[src]   (wave per edge, f32 atomics)
// ---------------------------------------------------------------------------
__global__ __launch_bounds__(256)
void edge_aggregate_kernel(const float* __restrict__ score, const float* __restrict__ den,
                           const __bf16* __restrict__ Vf, const int* __restrict__ src,
                           const int* __restrict__ dst, const int* __restrict__ etype,
                           float* __restrict__ out, int E)
{
  const int e    = (blockIdx.x * blockDim.x + threadIdx.x) >> 5;
  const int lane = threadIdx.x & 31;
  if (e >= E) return;
  const int s = src[e], d = dst[e], et = etype[e];
  const int head = lane >> 3;
  float attn = score[(size_t)e * 4 + head] / den[(size_t)(d * 4 + et) * 4 + head];
  v4bf v = *(const v4bf*)(Vf + (size_t)s * DMODEL + lane * 4);
  float* o = out + (size_t)d * DMODEL + lane * 4;
  atomicAdd(o + 0, attn * (float)v[0]);
  atomicAdd(o + 1, attn * (float)v[1]);
  atomicAdd(o + 2, attn * (float)v[2]);
  atomicAdd(o + 3, attn * (float)v[3]);
}

// ---------------------------------------------------------------------------
// h = LayerNorm(h + r) * g + b  — one wave per 128-wide row. Emits bf16 copy.
// ---------------------------------------------------------------------------
__global__ __launch_bounds__(256)
void add_ln_kernel(float* __restrict__ h, __bf16* __restrict__ hb, const float* __restrict__ r,
                   const float* __restrict__ g, const float* __restrict__ b, int n)
{
  const int row  = (blockIdx.x * blockDim.x + threadIdx.x) >> 5;
  const int lane = threadIdx.x & 31;
  if (row >= n) return;
  float4 hv = *(const float4*)(h + (size_t)row * DMODEL + lane * 4);
  float4 rv = *(const float4*)(r + (size_t)row * DMODEL + lane * 4);
  float x0 = hv.x + rv.x, x1 = hv.y + rv.y, x2 = hv.z + rv.z, x3 = hv.w + rv.w;
  float s = x0 + x1 + x2 + x3;
  #pragma unroll
  for (int m = 1; m < 32; m <<= 1) s += __shfl_xor(s, m);
  float mean = s * (1.0f / 128.0f);
  float q = (x0-mean)*(x0-mean) + (x1-mean)*(x1-mean) + (x2-mean)*(x2-mean) + (x3-mean)*(x3-mean);
  #pragma unroll
  for (int m = 1; m < 32; m <<= 1) q += __shfl_xor(q, m);
  float inv = rsqrtf(q * (1.0f / 128.0f) + 1e-5f);
  int c = lane * 4;
  float y0 = (x0 - mean) * inv * g[c + 0] + b[c + 0];
  float y1 = (x1 - mean) * inv * g[c + 1] + b[c + 1];
  float y2 = (x2 - mean) * inv * g[c + 2] + b[c + 2];
  float y3 = (x3 - mean) * inv * g[c + 3] + b[c + 3];
  float* o = h + (size_t)row * DMODEL + c;
  o[0] = y0; o[1] = y1; o[2] = y2; o[3] = y3;
  v4bf ob; ob[0] = (__bf16)y0; ob[1] = (__bf16)y1; ob[2] = (__bf16)y2; ob[3] = (__bf16)y3;
  *(v4bf*)(hb + (size_t)row * DMODEL + c) = ob;
}

// ---------------------------------------------------------------------------
// BatchNorm (training stats): per-channel block partials -> atomics.
// ---------------------------------------------------------------------------
__global__ __launch_bounds__(128)
void bn_stats_kernel(const float* __restrict__ r, float* __restrict__ sum,
                     float* __restrict__ sumsq, int n)
{
  const int c  = threadIdx.x;
  const int r0 = blockIdx.x * 128;
  float s = 0.0f, q = 0.0f;
  for (int i = 0; i < 128; ++i) {
    int row = r0 + i;
    if (row < n) { float v = r[(size_t)row * DMODEL + c]; s += v; q += v * v; }
  }
  atomicAdd(&sum[c], s);
  atomicAdd(&sumsq[c], q);
}

__global__ __launch_bounds__(128)
void bn_finalize_kernel(const float* __restrict__ sum, const float* __restrict__ sumsq,
                        const float* __restrict__ g, const float* __restrict__ b,
                        float* __restrict__ scale, float* __restrict__ shift, int n)
{
  const int c = threadIdx.x;
  float mean = sum[c] / (float)n;
  float var  = sumsq[c] / (float)n - mean * mean;
  float sc   = g[c] * rsqrtf(var + 1e-5f);
  scale[c] = sc;
  shift[c] = b[c] - mean * sc;
}

__global__ void bn_apply_kernel(const float* __restrict__ r, const float* __restrict__ scale,
                                const float* __restrict__ shift, float* __restrict__ h,
                                __bf16* __restrict__ hb, int total)
{
  int i = blockIdx.x * blockDim.x + threadIdx.x;
  if (i >= total) return;
  int c = i & (DMODEL - 1);
  float v = r[i] * scale[c] + shift[c];
  h[i]  = v;
  hb[i] = (__bf16)v;
}

// ---------------------------------------------------------------------------
// Attention pooling over all nodes.
// ---------------------------------------------------------------------------
__global__ __launch_bounds__(256)
void pool_score_kernel(const float* __restrict__ h, const float* __restrict__ q,
                       float* __restrict__ sbuf, unsigned* __restrict__ smax, int n)
{
  const int row  = (blockIdx.x * blockDim.x + threadIdx.x) >> 5;
  const int lane = threadIdx.x & 31;
  if (row >= n) return;
  float4 hv = *(const float4*)(h + (size_t)row * DMODEL + lane * 4);
  float4 qv = *(const float4*)(q + lane * 4);
  float p = hv.x * qv.x + hv.y * qv.y + hv.z * qv.z + hv.w * qv.w;
  #pragma unroll
  for (int m = 1; m < 32; m <<= 1) p += __shfl_xor(p, m);
  if (lane == 0) {
    sbuf[row] = p;
    atomicMax(smax, f2key(p));
  }
}

__global__ void pool_exp_kernel(float* __restrict__ sbuf, const unsigned* __restrict__ smax,
                                float* __restrict__ esum, int n)
{
  int i = blockIdx.x * blockDim.x + threadIdx.x;
  if (i >= n) return;
  float ex = expf(sbuf[i] - key2f(*smax));
  sbuf[i] = ex;
  atomicAdd(esum, ex);
}

__global__ __launch_bounds__(128)
void pool_accum_kernel(const float* __restrict__ h, const float* __restrict__ sbuf,
                       float* __restrict__ accum, int n)
{
  const int c  = threadIdx.x;
  const int r0 = blockIdx.x * 128;
  float a = 0.0f;
  for (int i = 0; i < 128; ++i) {
    int row = r0 + i;
    if (row < n) a += sbuf[row] * h[(size_t)row * DMODEL + c];
  }
  atomicAdd(&accum[c], a);
}

__global__ __launch_bounds__(128)
void pool_final_kernel(const float* __restrict__ accum, const float* __restrict__ esum,
                       float* __restrict__ out)
{
  out[threadIdx.x] = accum[threadIdx.x] / esum[0];
}

// ---------------------------------------------------------------------------
extern "C" void kernel_launch(void* const* d_in, const int* in_sizes, int n_in,
                              void* d_out, int out_size, void* d_ws, size_t ws_size,
                              hipStream_t stream)
{
  const float* x        = (const float*)d_in[0];
  const float* dist     = (const float*)d_in[1];
  const int*   src      = (const int*)  d_in[2];
  const int*   dstv     = (const int*)  d_in[3];
  const int*   etype    = (const int*)  d_in[4];
  const float* in_w     = (const float*)d_in[5];
  const float* in_b     = (const float*)d_in[6];
  const float* edge_emb = (const float*)d_in[7];
  const float* bias_w   = (const float*)d_in[8];
  const float* bias_b   = (const float*)d_in[9];
  const float* qw       = (const float*)d_in[10];
  const float* kw       = (const float*)d_in[11];
  const float* vw       = (const float*)d_in[12];
  const float* pw       = (const float*)d_in[13];
  const float* pb       = (const float*)d_in[14];
  const float* ln1g     = (const float*)d_in[15];
  const float* ln1b     = (const float*)d_in[16];
  const float* ln2g     = (const float*)d_in[17];
  const float* ln2b     = (const float*)d_in[18];
  const float* fw1      = (const float*)d_in[19];
  const float* fb1      = (const float*)d_in[20];
  const float* fw2      = (const float*)d_in[21];
  const float* fb2      = (const float*)d_in[22];
  const float* postw    = (const float*)d_in[23];
  const float* postb    = (const float*)d_in[24];
  const float* bng      = (const float*)d_in[25];
  const float* bnb      = (const float*)d_in[26];
  const float* poolq    = (const float*)d_in[27];
  float* out = (float*)d_out;

  // ---- workspace carve-up (256B aligned) ----
  char* wsb = (char*)d_ws;
  size_t off = 0;
  auto carve = [&](size_t bytes) -> void* {
    void* p = wsb + off;
    off = (off + bytes + 255) & ~(size_t)255;
    return p;
  };
  float*    h      = (float*)   carve(sizeof(float)  * (size_t)NNODE * DMODEL);
  __bf16*   hb     = (__bf16*)  carve(sizeof(__bf16) * (size_t)NNODE * DMODEL);
  __bf16*   bqb    = (__bf16*)  carve(sizeof(__bf16) * (size_t)NNODE * DMODEL);
  __bf16*   bkb    = (__bf16*)  carve(sizeof(__bf16) * (size_t)NNODE * DMODEL);
  __bf16*   bvb    = (__bf16*)  carve(sizeof(__bf16) * (size_t)NNODE * DMODEL);
  float*    battn  = (float*)   carve(sizeof(float)  * (size_t)NNODE * DMODEL);
  __bf16*   battnb = (__bf16*)  carve(sizeof(__bf16) * (size_t)NNODE * DMODEL);
  float*    bt     = (float*)   carve(sizeof(float)  * (size_t)NNODE * DMODEL);
  __bf16*   bfb    = (__bf16*)  carve(sizeof(__bf16) * (size_t)NNODE * FFDIM);
  float*    ebias  = (float*)   carve(sizeof(float)  * (size_t)NEDGE);
  float*    score  = (float*)   carve(sizeof(float)  * (size_t)NEDGE * 4);
  unsigned* mkey   = (unsigned*)carve(sizeof(unsigned) * (size_t)NNODE * 16);
  float*    den    = (float*)   carve(sizeof(float)    * (size_t)NNODE * 16);
  float*    bnsum  = (float*)   carve(sizeof(float) * DMODEL);
  float*    bnsq   = (float*)   carve(sizeof(float) * DMODEL);
  float*    bnsc   = (float*)   carve(sizeof(float) * DMODEL);
  float*    bnsh   = (float*)   carve(sizeof(float) * DMODEL);
  float*    sbuf   = (float*)   carve(sizeof(float) * (size_t)NNODE);
  unsigned* smax   = (unsigned*)carve(sizeof(unsigned));
  float*    esum   = (float*)   carve(sizeof(float));
  float*    paccu  = (float*)   carve(sizeof(float) * DMODEL);
  // bf16 mirrors of x and all GEMM weights (converted once)
  __bf16*   xb     = (__bf16*)  carve(sizeof(__bf16) * (size_t)NNODE * IN_DIM);
  __bf16*   wib    = (__bf16*)  carve(sizeof(__bf16) * (size_t)DMODEL * IN_DIM);
  __bf16*   wqb    = (__bf16*)  carve(sizeof(__bf16) * (size_t)NLAYER * DMODEL * DMODEL);
  __bf16*   wkb    = (__bf16*)  carve(sizeof(__bf16) * (size_t)NLAYER * DMODEL * DMODEL);
  __bf16*   wvb    = (__bf16*)  carve(sizeof(__bf16) * (size_t)NLAYER * DMODEL * DMODEL);
  __bf16*   wpb    = (__bf16*)  carve(sizeof(__bf16) * (size_t)NLAYER * DMODEL * DMODEL);
  __bf16*   wf1b   = (__bf16*)  carve(sizeof(__bf16) * (size_t)NLAYER * FFDIM * DMODEL);
  __bf16*   wf2b   = (__bf16*)  carve(sizeof(__bf16) * (size_t)NLAYER * DMODEL * FFDIM);
  __bf16*   wpostb = (__bf16*)  carve(sizeof(__bf16) * (size_t)NLAYER * DMODEL * DMODEL);
  (void)ws_size; (void)in_sizes; (void)n_in; (void)out_size;

  auto cvt = [&](const float* in, __bf16* o, size_t n) {
    f32_to_bf16_kernel<<<(unsigned)((n / 4 + 255) / 256), 256, 0, stream>>>(in, o, n);
  };

  const int rowWaveBlk  = (NNODE / 16 + 7) / 8;        // NT=8 GEMM / CG=1
  const int gemmD_blk   = (NNODE / 16 * 2 + 7) / 8;    // NC=128, NT=4 -> CG=2
  const int gemmF_blk   = (NNODE / 16 * 8 + 7) / 8;    // NC=512, NT=4 -> CG=8
  const int edgeWaveBlk = (NEDGE + 7) / 8;
  const int rowWave     = (NNODE + 7) / 8;
  const int rowChunkBlk = (NNODE + 127) / 128;

  // ---- one-time bf16 mirrors ----
  cvt(x, xb, (size_t)NNODE * IN_DIM);
  cvt(in_w, wib, (size_t)DMODEL * IN_DIM);
  cvt(qw, wqb, (size_t)NLAYER * DMODEL * DMODEL);
  cvt(kw, wkb, (size_t)NLAYER * DMODEL * DMODEL);
  cvt(vw, wvb, (size_t)NLAYER * DMODEL * DMODEL);
  cvt(pw, wpb, (size_t)NLAYER * DMODEL * DMODEL);
  cvt(fw1, wf1b, (size_t)NLAYER * FFDIM * DMODEL);
  cvt(fw2, wf2b, (size_t)NLAYER * DMODEL * FFDIM);
  cvt(postw, wpostb, (size_t)NLAYER * DMODEL * DMODEL);

  // ---- edge bias (stop-gradient path) ----
  edge_bias_kernel<<<(NEDGE + 255) / 256, 256, 0, stream>>>(
      dist, etype, edge_emb, bias_w, bias_b, ebias, NEDGE);

  // ---- input projection: h = x @ in_w^T + in_b (f32 + bf16 outputs) ----
  gemm_wmma_kernel<IN_DIM, DMODEL, 8, ACT_NONE><<<rowWaveBlk, 256, 0, stream>>>(
      xb, wib, in_b, h, hb, NNODE);

  for (int l = 0; l < NLAYER; ++l) {
    const size_t oDD = (size_t)l * DMODEL * DMODEL;

    // Q/K/V projections (no bias in reference); bf16 outputs only
    gemm_wmma_kernel<DMODEL, DMODEL, 4, ACT_NONE><<<gemmD_blk, 256, 0, stream>>>(
        hb, wqb + oDD, nullptr, nullptr, bqb, NNODE);
    gemm_wmma_kernel<DMODEL, DMODEL, 4, ACT_NONE><<<gemmD_blk, 256, 0, stream>>>(
        hb, wkb + oDD, nullptr, nullptr, bkb, NNODE);
    gemm_wmma_kernel<DMODEL, DMODEL, 4, ACT_NONE><<<gemmD_blk, 256, 0, stream>>>(
        hb, wvb + oDD, nullptr, nullptr, bvb, NNODE);

    hipMemsetAsync(mkey,  0, sizeof(unsigned) * (size_t)NNODE * 16, stream);
    hipMemsetAsync(den,   0, sizeof(float)    * (size_t)NNODE * 16, stream);
    hipMemsetAsync(battn, 0, sizeof(float)    * (size_t)NNODE * DMODEL, stream);

    edge_score_kernel<<<edgeWaveBlk, 256, 0, stream>>>(
        bkb, bqb, ebias, src, dstv, etype, score, mkey, NEDGE);
    edge_softmax_kernel<<<(NEDGE * 4 + 255) / 256, 256, 0, stream>>>(
        score, mkey, den, dstv, etype, NEDGE);
    edge_aggregate_kernel<<<edgeWaveBlk, 256, 0, stream>>>(
        score, den, bvb, src, dstv, etype, battn, NEDGE);

    // output projection + residual + LN1
    cvt(battn, battnb, (size_t)NNODE * DMODEL);
    gemm_wmma_kernel<DMODEL, DMODEL, 4, ACT_NONE><<<gemmD_blk, 256, 0, stream>>>(
        battnb, wpb + oDD, pb + (size_t)l * DMODEL, bt, nullptr, NNODE);
    add_ln_kernel<<<rowWave, 256, 0, stream>>>(
        h, hb, bt, ln1g + (size_t)l * DMODEL, ln1b + (size_t)l * DMODEL, NNODE);

    // FFN: gelu(h@w1^T+b1)@w2^T+b2, residual + LN2 (intermediate bf16-only)
    gemm_wmma_kernel<DMODEL, FFDIM, 4, ACT_GELU><<<gemmF_blk, 256, 0, stream>>>(
        hb, wf1b + (size_t)l * FFDIM * DMODEL, fb1 + (size_t)l * FFDIM, nullptr, bfb, NNODE);
    gemm_wmma_kernel<FFDIM, DMODEL, 4, ACT_NONE><<<gemmD_blk, 256, 0, stream>>>(
        bfb, wf2b + (size_t)l * DMODEL * FFDIM, fb2 + (size_t)l * DMODEL, bt, nullptr, NNODE);
    add_ln_kernel<<<rowWave, 256, 0, stream>>>(
        h, hb, bt, ln2g + (size_t)l * DMODEL, ln2b + (size_t)l * DMODEL, NNODE);

    // post linear -> relu -> batchnorm (training stats)
    gemm_wmma_kernel<DMODEL, DMODEL, 4, ACT_RELU><<<gemmD_blk, 256, 0, stream>>>(
        hb, wpostb + oDD, postb + (size_t)l * DMODEL, bt, nullptr, NNODE);
    hipMemsetAsync(bnsum, 0, sizeof(float) * DMODEL, stream);
    hipMemsetAsync(bnsq,  0, sizeof(float) * DMODEL, stream);
    bn_stats_kernel<<<rowChunkBlk, 128, 0, stream>>>(bt, bnsum, bnsq, NNODE);
    bn_finalize_kernel<<<1, 128, 0, stream>>>(bnsum, bnsq, bng + (size_t)l * DMODEL,
                                              bnb + (size_t)l * DMODEL, bnsc, bnsh, NNODE);
    bn_apply_kernel<<<(NNODE * DMODEL + 255) / 256, 256, 0, stream>>>(
        bt, bnsc, bnsh, h, hb, NNODE * DMODEL);
  }

  // ---- attention pooling ----
  hipMemsetAsync(smax,  0, sizeof(unsigned), stream);
  hipMemsetAsync(esum,  0, sizeof(float), stream);
  hipMemsetAsync(paccu, 0, sizeof(float) * DMODEL, stream);
  pool_score_kernel<<<rowWave, 256, 0, stream>>>(h, poolq, sbuf, smax, NNODE);
  pool_exp_kernel<<<(NNODE + 255) / 256, 256, 0, stream>>>(sbuf, smax, esum, NNODE);
  pool_accum_kernel<<<rowChunkBlk, 128, 0, stream>>>(h, sbuf, paccu, NNODE);
  pool_final_kernel<<<1, 128, 0, stream>>>(paccu, esum, out);
}
